// ManualConv2d_40321152975166
// MI455X (gfx1250) — compile-verified, matrix operand order checked by reading, not run on hardware
//
#include <hip/hip_runtime.h>

// ---- problem constants (from reference) ----
constexpr int CN   = 32;        // batch
constexpr int CIN  = 128;       // input channels
constexpr int CH   = 56;
constexpr int CW   = 56;
constexpr int COUT = 256;
constexpr int HW   = CH * CW;   // 3136
constexpr int NPIX = CN * HW;   // 100352

// ---- tiling ----
#define TM     128              // C_out tile
#define TN     128              // pixel tile
#define TK     32               // K chunk (bf16 WMMA K)
#define LDSS32 20               // padded LDS row stride in u32 (== 40 bf16, 80 B)
#define NSTEP  36               // 9 taps * 4 ci-chunks

typedef __attribute__((ext_vector_type(16))) __bf16 v16bf;
typedef __attribute__((ext_vector_type(2)))  __bf16 v2bf;
typedef __attribute__((ext_vector_type(8)))  float  v8f;

struct B32x8 { uint4 lo, hi; };   // 32 bytes == v16bf

// pack two f32 -> two bf16 in one u32 (a -> low half, b -> high half).
// Round-to-nearest (ties away) then a single v_perm_b32 merges both high halves:
// 3 VALU ops per pair instead of a 5-op RNE chain.
__device__ __forceinline__ unsigned pack_bf16(float a, float b) {
#if __has_builtin(__builtin_amdgcn_cvt_pk_bf16_f32)
    v2bf r = __builtin_amdgcn_cvt_pk_bf16_f32(a, b);   // v_cvt_pk_bf16_f32
    return __builtin_bit_cast(unsigned, r);
#else
    unsigned ua = __builtin_bit_cast(unsigned, a) + 0x8000u;
    unsigned ub = __builtin_bit_cast(unsigned, b) + 0x8000u;
    // dst bytes {ub[3],ub[2],ua[3],ua[2]}  (S0=ub -> idx 4..7, S1=ua -> idx 0..3)
    return __builtin_amdgcn_perm(ub, ua, 0x07060302u);
#endif
}

__global__ __launch_bounds__(256, 2)
void conv2d_wmma_kernel(const float* __restrict__ in,
                        const float* __restrict__ wt,
                        const float* __restrict__ bias,
                        float* __restrict__ out)
{
    // ping-pong staged tiles (bf16 packed as u32): total 40 KB LDS
    __shared__ __align__(16) unsigned As[2][TM * LDSS32]; // weights [co][ci]
    __shared__ __align__(16) unsigned Bs[2][TN * LDSS32]; // pixels  [px][ci]

    const int tid  = threadIdx.x;
    const int lane = tid & 31;
    const int wave = tid >> 5;
    const int wm   = wave & 3;      // 4 waves along M: 32 rows each
    const int wn   = wave >> 2;     // 2 waves along N: 64 pixels each
    const int l16  = lane & 15;
    const int half = lane >> 4;     // K-half select per 16-bit A/B layout

    const int mtile    = blockIdx.x & 1;
    const int ntile    = blockIdx.x >> 1;
    const int co_base  = mtile * TM;
    const int pix_base = ntile * TN;

    // ---- loop-invariant staging geometry (hoisted out of the K loop) ----
    // B side: each thread owns one pixel and ci pairs {2*cipB + 4*i, +1}
    const int px   = tid & 127;
    const int cipB = tid >> 7;                  // 0..1
    const int p    = pix_base + px;
    const int n    = p / HW, hw = p - n * HW;
    const int h    = hw / CW, w = hw - h * CW;
    const int ibase = n * CIN * HW + h * CW + w;

    // A side: each thread owns one ci pair column and co rows {coA + 16*i}
    const int coA  = tid >> 4;                  // 0..15 (+16*i)
    const int cipA = tid & 15;
    int wbase[8];
    #pragma unroll
    for (int i = 0; i < 8; ++i)
        wbase[i] = ((co_base + coA + 16 * i) * CIN + 2 * cipA) * 9;

    v8f acc[2][4];
    #pragma unroll
    for (int i = 0; i < 2; ++i)
        #pragma unroll
        for (int j = 0; j < 4; ++j)
            acc[i][j] = v8f{};

    float av[16], bv[16];   // in-flight staging registers

    // issue all global loads for step s into av/bv (batched -> s_clause)
    auto stage_load = [&](int s) {
        const int tap = s >> 2, ccidx = s & 3;
        const int kh  = tap / 3, kw = tap - kh * 3;
        const int cc  = ccidx * TK;
        const int dh  = kh - 1, dw = kw - 1;
        const bool vld = ((unsigned)(h + dh) < (unsigned)CH) &&
                         ((unsigned)(w + dw) < (unsigned)CW);
        const int boff = ibase + (cc + 2 * cipB) * HW + dh * CW + dw;
        #pragma unroll
        for (int i = 0; i < 8; ++i) {
            const int a = boff + 4 * i * HW;
            bv[2 * i]     = vld ? in[a]      : 0.0f;
            bv[2 * i + 1] = vld ? in[a + HW] : 0.0f;
        }
        const int wo = cc * 9 + tap;
        #pragma unroll
        for (int i = 0; i < 8; ++i) {
            av[2 * i]     = wt[wbase[i] + wo];
            av[2 * i + 1] = wt[wbase[i] + wo + 9];
        }
    };

    // convert + packed ds_store_b32 into buffer `buf`
    auto stage_store = [&](int buf) {
        #pragma unroll
        for (int i = 0; i < 8; ++i) {
            Bs[buf][px * LDSS32 + cipB + 2 * i]     = pack_bf16(bv[2*i], bv[2*i+1]);
            As[buf][(coA + 16 * i) * LDSS32 + cipA] = pack_bf16(av[2*i], av[2*i+1]);
        }
    };

    // fragment loads (documented 16-bit A layout; B staged column-major) + 8 WMMAs
    auto compute = [&](int buf) {
        const unsigned short* A16 = (const unsigned short*)As[buf];
        const unsigned short* B16 = (const unsigned short*)Bs[buf];
        v16bf aF[2], bF[4];
        #pragma unroll
        for (int sm = 0; sm < 2; ++sm) {
            const unsigned short* pA = A16 + (wm * 32 + sm * 16 + l16) * (2 * LDSS32) + half * 8;
            B32x8 r; r.lo = *(const uint4*)pA; r.hi = *(const uint4*)(pA + 16);
            aF[sm] = __builtin_bit_cast(v16bf, r);
        }
        #pragma unroll
        for (int sn = 0; sn < 4; ++sn) {
            const unsigned short* pB = B16 + (wn * 64 + sn * 16 + l16) * (2 * LDSS32) + half * 8;
            B32x8 r; r.lo = *(const uint4*)pB; r.hi = *(const uint4*)(pB + 16);
            bF[sn] = __builtin_bit_cast(v16bf, r);
        }
        #pragma unroll
        for (int sm = 0; sm < 2; ++sm)
            #pragma unroll
            for (int sn = 0; sn < 4; ++sn)
                acc[sm][sn] = __builtin_amdgcn_wmma_f32_16x16x32_bf16(
                    false, aF[sm], false, bF[sn], (short)0, acc[sm][sn],
                    false, false);
    };

    // ---- software pipeline: 1 barrier / step, global loads overlap compute ----
    stage_load(0);
    stage_store(0);

    #pragma unroll 1
    for (int s = 0; s < NSTEP - 1; ++s) {
        __syncthreads();
        stage_load(s + 1);        // next tile's global loads in flight
        compute(s & 1);           // WMMA on current LDS buffer
        stage_store((s + 1) & 1); // park next tile into the other buffer
    }
    __syncthreads();
    compute((NSTEP - 1) & 1);

    // ---- epilogue: C/D layout (VGPR v -> M = v + 8*half, N = lane&15) ----
    // nontemporal: output is write-once; keep L2 for the re-read input
    #pragma unroll
    for (int sm = 0; sm < 2; ++sm) {
        const int co_sub = co_base + wm * 32 + sm * 16;
        #pragma unroll
        for (int sn = 0; sn < 4; ++sn) {
            const int pp = pix_base + wn * 64 + sn * 16 + l16;
            const int nn = pp / HW, hw2 = pp - nn * HW;
            #pragma unroll
            for (int v = 0; v < 8; ++v) {
                const int co = co_sub + v + half * 8;
                __builtin_nontemporal_store(acc[sm][sn][v] + bias[co],
                                            &out[(nn * COUT + co) * HW + hw2]);
            }
        }
    }
}

extern "C" void kernel_launch(void* const* d_in, const int* in_sizes, int n_in,
                              void* d_out, int out_size, void* d_ws, size_t ws_size,
                              hipStream_t stream) {
    (void)in_sizes; (void)n_in; (void)out_size; (void)d_ws; (void)ws_size;
    const float* in   = (const float*)d_in[0];
    const float* wt   = (const float*)d_in[1];
    const float* bias = (const float*)d_in[2];
    float* out        = (float*)d_out;

    const int grid = (COUT / TM) * (NPIX / TN);   // 2 * 784 = 1568 blocks
    conv2d_wmma_kernel<<<grid, 256, 0, stream>>>(in, wt, bias, out);
}
